// SpectralConvLayer_81913616269405
// MI455X (gfx1250) — matrix operand order repaired and checked by reading, not compiled, not run on hardware
//
#include <hip/hip_runtime.h>
#include <math.h>

// ---------------- problem constants (from reference setup_inputs) ------------
#define NB    16      // batch
#define CIN   64      // input channels  (K dim of GEMM)
#define COUT  64      // output channels (N dim of GEMM)
#define LLEN  8192    // sequence length
#define LOG2L 13
#define KLEN  64      // kernel length
#define PI_F  3.14159265358979323846f

typedef __attribute__((ext_vector_type(16))) _Float16 v16h;
typedef __attribute__((ext_vector_type(8)))  float    v8f;

struct h2   { _Float16 x, y; };
struct kent { h2 m; h2 p; };   // {mag_i0, mag_i1, ph_i0, ph_i1} packed, 8 B

#define WMMA_F16(A, B, C) \
  __builtin_amdgcn_wmma_f32_16x16x32_f16(false, (A), false, (B), (short)0, (C), false, false)

// ---------------- 1) per-batch min/max of |x| over (Cin, L) ------------------
// min/max of mag == sqrt of min/max of mag^2 (monotone) -> no sqrt in the loop
__global__ __launch_bounds__(256) void minmax_kernel(const float* __restrict__ xr,
                                                     const float* __restrict__ xi,
                                                     float2* __restrict__ mm) {
  int b = blockIdx.x;
  int t = threadIdx.x;
  const float* pr = xr + (size_t)b * CIN * LLEN;
  const float* pi = xi + (size_t)b * CIN * LLEN;
  float mn = 3.4e38f, mx = 0.0f;
  for (int idx = t; idx < CIN * LLEN; idx += 256) {
    float a = pr[idx], c = pi[idx];
    float m2 = a * a + c * c;
    mn = fminf(mn, m2); mx = fmaxf(mx, m2);
  }
  __shared__ float smn[256], smx[256];
  smn[t] = mn; smx[t] = mx;
  __syncthreads();
  for (int s = 128; s > 0; s >>= 1) {
    if (t < s) { smn[t] = fminf(smn[t], smn[t + s]); smx[t] = fmaxf(smx[t], smx[t + s]); }
    __syncthreads();
  }
  if (t == 0) mm[b] = make_float2(sqrtf(smn[0]), sqrtf(smx[0]));
}

// ---------------- 2) 64-pt DFT of kernels -> polar table (mag, phase) --------
__global__ __launch_bounds__(64) void kfft_kernel(const float* __restrict__ kr,
                                                  const float* __restrict__ ki,
                                                  float* __restrict__ kmag,
                                                  float* __restrict__ kph) {
  int row = blockIdx.x;          // o*CIN + i
  int k   = threadIdx.x;         // frequency bin
  const float* pr = kr + (size_t)row * KLEN;
  const float* pi = ki + (size_t)row * KLEN;
  float sr = 0.0f, si = 0.0f;
  for (int t = 0; t < KLEN; ++t) {
    // exact twiddle: e^{-2pi i k t / 64} is periodic mod 64
    float ang = -2.0f * PI_F * (float)((k * t) & (KLEN - 1)) * (1.0f / (float)KLEN);
    float s, c; __sincosf(ang, &s, &c);
    float ar = pr[t], ai = pi[t];
    sr += ar * c - ai * s;
    si += ar * s + ai * c;
  }
  size_t o = (size_t)row * KLEN + k;
  kmag[o] = sqrtf(sr * sr + si * si);
  kph[o]  = atan2f(si, sr);
}

// ---------------- 3) coupling + forward 8192-pt FFT (one row / block) --------
__global__ __launch_bounds__(256) void fft_fwd_kernel(const float* __restrict__ xr,
                                                      const float* __restrict__ xi,
                                                      const float2* __restrict__ mm,
                                                      float2* __restrict__ xfft) {
  __shared__ float2 s[LLEN];     // 64 KB (WGP has 320 KB LDS)
  int row = blockIdx.x;          // b*CIN + j
  int b   = row / CIN;
  int t   = threadIdx.x;
  float2 m = mm[b];
  float mn = m.x, mx = m.y;
  float inv = 1.0f / (mx - mn + 1e-10f);
  const float* pr = xr + (size_t)row * LLEN;
  const float* pi = xi + (size_t)row * LLEN;

  // load with logistic self-coupling applied; store bit-reversed
  for (int idx = t; idx < LLEN; idx += 256) {
    float a = pr[idx], c = pi[idx];
    float mag = sqrtf(a * a + c * c);
    float xn = (mag - mn) * inv;
#pragma unroll
    for (int it = 0; it < 5; ++it) xn = 3.8f * xn * (1.0f - xn);
    float magc = xn * (mx - mn) + mn;
    float sc = magc / (mag + 1e-30f);        // mag_c * e^{i phase} = x * (mag_c/mag)
    int r = (int)(__brev((unsigned)idx) >> (32 - LOG2L));
    s[r] = make_float2(a * sc, c * sc);
  }
  __syncthreads();

  // radix-2 DIT stages, forward sign (-)
  for (int st = 1; st <= LOG2L; ++st) {
    int mlen = 1 << st, mh = mlen >> 1;
    for (int w = t; w < (LLEN >> 1); w += 256) {
      int grp = w / mh, pos = w - grp * mh;
      int i = grp * mlen + pos, j = i + mh;
      float ang = -2.0f * PI_F * (float)pos / (float)mlen;
      float sn, cs; __sincosf(ang, &sn, &cs);
      float2 u = s[i], v = s[j];
      float2 vw = make_float2(v.x * cs - v.y * sn, v.x * sn + v.y * cs);
      s[i] = make_float2(u.x + vw.x, u.y + vw.y);
      s[j] = make_float2(u.x - vw.x, u.y - vw.y);
    }
    __syncthreads();
  }
  float2* out = xfft + (size_t)row * LLEN;
  for (int idx = t; idx < LLEN; idx += 256) out[idx] = s[idx];
}

// ---------------- 4) per-frequency complex GEMM via WMMA ---------------------
// out_fft[b,o,l] = sum_j x_fft[b,j,l] * kint[o,j,l]
// One wave32 per l; 8 waves/block share:
//   - an LDS-staged X tile [16x64] for 8 consecutive l's
//   - an LDS-staged interpolation-endpoint table (i0/i1 are identical for all
//     8 l's in a block: endpoints flip only at l = 128k+64, which is 8-aligned)
// Output is transposed through LDS for coalesced 64B global stores.
__global__ __launch_bounds__(256) void einsum_wmma_kernel(const float2* __restrict__ xfft,
                                                          const float* __restrict__ kmag,
                                                          const float* __restrict__ kph,
                                                          float2* __restrict__ offt) {
  __shared__ __align__(16) char smem[65536];      // phase-aliased 64 KB
  h2*     sX   = (h2*)smem;                       // [0,32K): [p=b*64+j][li], 8192
  kent*   sK   = (kent*)(smem + 32768);           // [32K,64K): 4096 entries (o*64+j)
  float2* sOut = (float2*)smem;                   // reused after compute: 8192

  int t  = threadIdx.x;
  int l0 = blockIdx.x * 8;

  // block-uniform interpolation endpoints
  float pos0 = ((float)l0 + 0.5f) * (1.0f / 128.0f) - 0.5f;
  pos0 = fminf(fmaxf(pos0, 0.0f), (float)(KLEN - 1));
  int i0 = (int)floorf(pos0);
  int i1 = i0 + 1; if (i1 > KLEN - 1) i1 = KLEN - 1;

  // pull the 2 MB polar table toward this WGP (fully reused across all l)
  __builtin_prefetch(kmag, 0, 3);
  __builtin_prefetch(kph,  0, 3);

  // ---- cooperative coalesced stage: X tile (8 consecutive l's) ----
  for (int f = t; f < NB * CIN * 8; f += 256) {
    int p = f >> 3, li = f & 7;
    float2 v = xfft[(size_t)p * LLEN + l0 + li];
    h2 e; e.x = (_Float16)v.x; e.y = (_Float16)v.y;
    sX[p * 8 + li] = e;
  }
  // ---- cooperative stage: interpolation endpoints for all (o,j) ----
  for (int q = t; q < COUT * CIN; q += 256) {
    int base = q * KLEN;
    kent e;
    e.m.x = (_Float16)kmag[base + i0];
    e.m.y = (_Float16)kmag[base + i1];
    e.p.x = (_Float16)kph [base + i0];
    e.p.y = (_Float16)kph [base + i1];
    sK[q] = e;
  }
  __syncthreads();

  int wave = t >> 5, lane = t & 31;   // wave32 on gfx1250
  int l = l0 + wave;
  int g = lane >> 4;                  // lane group (0: lanes 0-15, 1: lanes 16-31)
  int n = lane & 15;                  // N column within tile
  int m = lane & 15;                  // M row for A

  // per-l interpolation weight against the block-uniform endpoints
  float pos = ((float)l + 0.5f) * (1.0f / 128.0f) - 0.5f;
  pos = fminf(fmaxf(pos, 0.0f), (float)(KLEN - 1));
  float w = pos - (float)i0;

  // ---- pack A (16x64 as two 16x32 f16 chunks) in exact ISA VGPR layout ----
  v16h Ar0, Ai0, Ar1, Ai1;
#pragma unroll
  for (int h = 0; h < 16; ++h) {
    int K0 = 16 * (h >> 3) + 8 * g + (((h >> 1) & 3) << 1) + (h & 1);
    h2 e0 = sX[(m * CIN + K0) * 8 + wave];
    h2 e1 = sX[(m * CIN + K0 + 32) * 8 + wave];
    Ar0[h] = e0.x; Ai0[h] = e0.y;
    Ar1[h] = e1.x; Ai1[h] = e1.y;
  }

  // ---- 4 N-tiles of 16 output channels each; keep all C in registers ----
  v8f Cr[4], Ci[4];
#pragma unroll
  for (int tile = 0; tile < 4; ++tile) {
    int o = tile * 16 + n;
    v16h Br0, Bi0, Bn0, Br1, Bi1, Bn1;   // Bn = -Bi (avoids A-neg modifier)
#pragma unroll
    for (int h = 0; h < 16; ++h) {
      int j0 = g * 8 + h + 8 * (h >> 3); // B layout: K = 8*g + h (+8 for h>=8), N = lane%16
      kent e0 = sK[o * CIN + j0];
      float mg0 = (float)e0.m.x + ((float)e0.m.y - (float)e0.m.x) * w;
      float ph0 = (float)e0.p.x + ((float)e0.p.y - (float)e0.p.x) * w;
      float sn0, cs0; __sincosf(ph0, &sn0, &cs0);
      Br0[h] = (_Float16)(mg0 * cs0);
      Bi0[h] = (_Float16)(mg0 * sn0);
      Bn0[h] = (_Float16)(-mg0 * sn0);

      kent e1 = sK[o * CIN + j0 + 32];
      float mg1 = (float)e1.m.x + ((float)e1.m.y - (float)e1.m.x) * w;
      float ph1 = (float)e1.p.x + ((float)e1.p.y - (float)e1.p.x) * w;
      float sn1, cs1; __sincosf(ph1, &sn1, &cs1);
      Br1[h] = (_Float16)(mg1 * cs1);
      Bi1[h] = (_Float16)(mg1 * sn1);
      Bn1[h] = (_Float16)(-mg1 * sn1);
    }

    v8f cr = {}; v8f ci = {};
    // Cr = Ar*Br - Ai*Bi   (chained f32 accumulation)
    cr = WMMA_F16(Ar0, Br0, cr);
    cr = WMMA_F16(Ar1, Br1, cr);
    cr = WMMA_F16(Ai0, Bn0, cr);
    cr = WMMA_F16(Ai1, Bn1, cr);
    // Ci = Ar*Bi + Ai*Br
    ci = WMMA_F16(Ar0, Bi0, ci);
    ci = WMMA_F16(Ar1, Bi1, ci);
    ci = WMMA_F16(Ai0, Br0, ci);
    ci = WMMA_F16(Ai1, Br1, ci);
    Cr[tile] = cr; Ci[tile] = ci;
  }

  __syncthreads();   // all waves done reading sX/sK -> safe to alias as sOut

  // scatter C into LDS: C/D layout VGPR v -> M = v + 8*g, N = lane%16
#pragma unroll
  for (int tile = 0; tile < 4; ++tile) {
#pragma unroll
    for (int v = 0; v < 8; ++v) {
      int p = (v + 8 * g) * COUT + tile * 16 + n;   // b*COUT + o
      sOut[p * 8 + wave] = make_float2(Cr[tile][v], Ci[tile][v]);
    }
  }
  __syncthreads();

  // cooperative coalesced store: 8 threads -> 64 contiguous bytes per row
  for (int f = t; f < NB * COUT * 8; f += 256) {
    int p = f >> 3, li = f & 7;
    offt[(size_t)p * LLEN + l0 + li] = sOut[f];
  }
}

// ---------------- 5) inverse FFT + sin(alpha*l) activation -------------------
__global__ __launch_bounds__(256) void ifft_act_kernel(const float2* __restrict__ offt,
                                                       const float* __restrict__ alpha,
                                                       float* __restrict__ out) {
  __shared__ float2 s[LLEN];     // 64 KB
  int row = blockIdx.x;          // b*COUT + o
  int t   = threadIdx.x;
  const float2* in = offt + (size_t)row * LLEN;
  for (int idx = t; idx < LLEN; idx += 256) {
    int r = (int)(__brev((unsigned)idx) >> (32 - LOG2L));
    s[r] = in[idx];
  }
  __syncthreads();
  for (int st = 1; st <= LOG2L; ++st) {
    int mlen = 1 << st, mh = mlen >> 1;
    for (int w = t; w < (LLEN >> 1); w += 256) {
      int grp = w / mh, pos = w - grp * mh;
      int i = grp * mlen + pos, j = i + mh;
      float ang = 2.0f * PI_F * (float)pos / (float)mlen;   // inverse sign
      float sn, cs; __sincosf(ang, &sn, &cs);
      float2 u = s[i], v = s[j];
      float2 vw = make_float2(v.x * cs - v.y * sn, v.x * sn + v.y * cs);
      s[i] = make_float2(u.x + vw.x, u.y + vw.y);
      s[j] = make_float2(u.x - vw.x, u.y - vw.y);
    }
    __syncthreads();
  }
  float al = alpha[0];
  const float invL = 1.0f / (float)LLEN;
  float* o = out + (size_t)row * LLEN * 2;   // interleaved complex64
  for (int idx = t; idx < LLEN; idx += 256) {
    float act = __sinf(al * (float)idx) * invL;  // ifft 1/L folded into activation
    o[2 * idx + 0] = s[idx].x * act;
    o[2 * idx + 1] = s[idx].y * act;
  }
}

// ---------------- launcher ---------------------------------------------------
extern "C" void kernel_launch(void* const* d_in, const int* in_sizes, int n_in,
                              void* d_out, int out_size, void* d_ws, size_t ws_size,
                              hipStream_t stream) {
  const float* xr    = (const float*)d_in[0];
  const float* xi    = (const float*)d_in[1];
  const float* kr    = (const float*)d_in[2];
  const float* ki    = (const float*)d_in[3];
  const float* alpha = (const float*)d_in[4];

  char* ws = (char*)d_ws;
  size_t off = 0;
  float2* mm   = (float2*)(ws + off); off += 256;                                  // 16 x float2
  float*  kmag = (float*)(ws + off);  off += (size_t)COUT * CIN * KLEN * 4;        // 1 MB
  float*  kph  = (float*)(ws + off);  off += (size_t)COUT * CIN * KLEN * 4;        // 1 MB
  float2* xfft = (float2*)(ws + off); off += (size_t)NB * CIN  * LLEN * 8;         // 67 MB
  float2* offt = (float2*)(ws + off); off += (size_t)NB * COUT * LLEN * 8;         // 67 MB
  float*  out  = (float*)d_out;

  minmax_kernel     <<<NB,          256, 0, stream>>>(xr, xi, mm);
  kfft_kernel       <<<COUT * CIN,  64,  0, stream>>>(kr, ki, kmag, kph);
  fft_fwd_kernel    <<<NB * CIN,    256, 0, stream>>>(xr, xi, mm, xfft);
  einsum_wmma_kernel<<<LLEN / 8,    256, 0, stream>>>(xfft, kmag, kph, offt);
  ifft_act_kernel   <<<NB * COUT,   256, 0, stream>>>(offt, alpha, out);
}
